// MultiHead_Self_Attention_42984032698704
// MI455X (gfx1250) — matrix operand order repaired and checked by reading, not compiled, not run on hardware
//
#include <hip/hip_runtime.h>
#include <hip/hip_bf16.h>

// ---------------------------------------------------------------------------
// MultiHead self-attention (b=4, dim=512, L=2048, H=8, Dh=64) on gfx1250.
// f16 WMMA (v_wmma_f32_16x16x32_f16) with fp32 accumulation everywhere.
// Flash-attention stage 2 (online softmax) so scores never touch HBM.
// Round 3: branchless strided epilogue in kernel 1 (kills 32 per-store
// s_cbranch diamonds seen in round-2 asm); rest unchanged.
// ---------------------------------------------------------------------------

typedef __attribute__((ext_vector_type(16))) _Float16 v16h;
typedef __attribute__((ext_vector_type(8)))  _Float16 half8;
typedef __attribute__((ext_vector_type(2)))  _Float16 h2;
typedef __attribute__((ext_vector_type(8)))  float    v8f;

#define HEADS    8
#define DHEAD    64
#define SEQ      2048
#define DIM      512
#define OQKV     1536
#define BATCH    4
#define ATT_SCALE 0.125f   // 64^-0.5

__device__ __forceinline__ v8f wmma_f16(v16h a, v16h b, v8f c) {
  // (neg_a, A, neg_b, B, c_mod, C, reuse_a, reuse_b)
  return __builtin_amdgcn_wmma_f32_16x16x32_f16(false, a, false, b, (short)0, c,
                                                false, false);
}

// A-matrix fragment (16x32 f16), ISA 7.12.2:
//   lane m = row (lane&15); half g: e<8 -> k = kb+8g+e, e>=8 -> k = kb+16+8g+(e-8)
__device__ __forceinline__ v16h load_a_frag(const _Float16* base, int stride,
                                            int row, int g, int kbase) {
  const _Float16* p = base + (size_t)row * stride + kbase + 8 * g;
  half8 lo = *(const half8*)(p);
  half8 hi = *(const half8*)(p + 16);
  v16h r;
#pragma unroll
  for (int e = 0; e < 8; ++e) { r[e] = lo[e]; r[e + 8] = hi[e]; }
  return r;
}

// B-matrix fragment (32x16 f16): lane n = col (lane&15), same k order as A:
// e<8 -> k = 8g+e, e>=8 -> k = 16+8g+(e-8); source stored [n][k] row-major.
__device__ __forceinline__ v16h load_b_frag(const _Float16* base, int stride,
                                            int col, int g) {
  const _Float16* p = base + (size_t)col * stride + 8 * g;
  half8 lo = *(const half8*)(p);
  half8 hi = *(const half8*)(p + 16);
  v16h r;
#pragma unroll
  for (int e = 0; e < 8; ++e) { r[e] = lo[e]; r[e + 8] = hi[e]; }
  return r;
}

// ---------------------------------------------------------------------------
// Kernel 1: QKV projection.  C[o,l] = W[o,:] . X[b,:,l] + bias, o in [0,1536)
// Block = 128 threads (4 waves), tile = 64(o) x 64(l), K-step = 32.
// Q (pre-scaled) and K written f16 [b][h][l][64]; V written f16 [b][h][64][l].
// ---------------------------------------------------------------------------
__global__ __launch_bounds__(128) void qkv_proj_kernel(
    const float* __restrict__ X, const float* __restrict__ W,
    const float* __restrict__ Bias, _Float16* __restrict__ Qb,
    _Float16* __restrict__ Kb, _Float16* __restrict__ Vb) {
  const int b  = blockIdx.z;
  const int o0 = blockIdx.y * 64;
  const int l0 = blockIdx.x * 64;
  const int tid  = threadIdx.x;
  const int lane = tid & 31, w = tid >> 5;
  const int g = lane >> 4, ln = lane & 15;

  __shared__ _Float16 sW[64 * 32];   // [o_local][c_local]
  __shared__ _Float16 sXT[64 * 32];  // [l_local][c_local]  (transposed)

  v8f acc[4];
#pragma unroll
  for (int nt = 0; nt < 4; ++nt)
#pragma unroll
    for (int e = 0; e < 8; ++e) acc[nt][e] = 0.0f;

  for (int c0 = 0; c0 < DIM; c0 += 32) {
    __syncthreads();
    {  // stage W tile, f32 -> f16, packed b32 LDS stores
      int idx = tid * 16, o = idx >> 5, cs = idx & 31;
      const float* src = W + (size_t)(o0 + o) * DIM + c0 + cs;
#pragma unroll
      for (int e = 0; e < 8; ++e) {
        h2 hv = {(_Float16)src[2 * e], (_Float16)src[2 * e + 1]};
        *(h2*)(sW + o * 32 + cs + 2 * e) = hv;
      }
    }
    {  // stage X tile transposed: two c-rows per thread -> packed [l][c] pairs
      int cp = tid >> 3;            // c-pair 0..15
      int ls = (tid & 7) * 8;       // l start
      const float* s0 = X + ((size_t)b * DIM + c0 + 2 * cp) * SEQ + l0 + ls;
      const float* s1 = s0 + SEQ;
#pragma unroll
      for (int e = 0; e < 8; ++e) {
        h2 hv = {(_Float16)s0[e], (_Float16)s1[e]};
        *(h2*)(sXT + (ls + e) * 32 + 2 * cp) = hv;
      }
    }
    __syncthreads();
    v16h af = load_a_frag(sW, 32, w * 16 + ln, g, 0);
#pragma unroll
    for (int nt = 0; nt < 4; ++nt) {
      v16h bf = load_b_frag(sXT, 32, nt * 16 + ln, g);
      acc[nt] = wmma_f16(af, bf, acc[nt]);
    }
  }

  // Epilogue: C layout -> lane holds col n = ln (==l), rows m = r + 8g.
  // Branchless layout select: Q/K -> [l][d] (sd=1, sl=64); V -> [d][l]
  // (sd=SEQ, sl=1).  Uniform scalars, no per-store control flow.
  const int sec = o0 / DIM;             // 0=q 1=k 2=v (o0 is 64-aligned)
  const int h   = (o0 % DIM) / DHEAD;   // uniform per block
  _Float16* dst = (sec == 0 ? Qb : (sec == 1 ? Kb : Vb)) +
                  (size_t)(b * HEADS + h) * SEQ * DHEAD;
  const float scale = (sec == 0) ? ATT_SCALE : 1.0f;
  const int sd = (sec == 2) ? SEQ : 1;
  const int sl = (sec == 2) ? 1 : DHEAD;
#pragma unroll
  for (int r = 0; r < 8; ++r) {
    const int d    = 16 * w + r + 8 * g;     // o within this 64-row tile
    const float bi = Bias[o0 + d];
#pragma unroll
    for (int nt = 0; nt < 4; ++nt) {
      const int l = l0 + nt * 16 + ln;
      dst[(size_t)d * sd + (size_t)l * sl] =
          (_Float16)((acc[nt][r] + bi) * scale);
    }
  }
}

// ---------------------------------------------------------------------------
// Kernel 2: flash attention per (b,h).  Block = 4 waves = 64 query rows;
// each wave owns 16 rows (Q fragments resident), K/V tiles (32 keys) shared
// via LDS.  P stored pair-packed ((j, j+16) as one dword) so both LDS sides
// are wide accesses.
// ---------------------------------------------------------------------------
__global__ __launch_bounds__(128) void flash_attn_kernel(
    const _Float16* __restrict__ Qb, const _Float16* __restrict__ Kb,
    const _Float16* __restrict__ Vb, _Float16* __restrict__ Ob) {
  const int bh  = blockIdx.z * HEADS + blockIdx.y;
  const int tid = threadIdx.x;
  const int lane = tid & 31, w = tid >> 5;
  const int g = lane >> 4, ln = lane & 15;
  const int i0 = blockIdx.x * 64 + w * 16;

  __shared__ _Float16 sK[32 * 64];      // [j][d]
  __shared__ _Float16 sVT[64 * 32];     // [d][j]  (already transposed in HBM)
  __shared__ _Float16 sP[4][16 * 32];   // per-wave, pair-interleaved [i][(j,j+16)]

  // Resident Q A-fragments (16 rows x 64 d, two k-chunks of 32)
  const _Float16* qbase = Qb + ((size_t)bh * SEQ + i0) * DHEAD;
  v16h qa[2];
#pragma unroll
  for (int kc = 0; kc < 2; ++kc) qa[kc] = load_a_frag(qbase, DHEAD, ln, g, 32 * kc);

  v8f oacc[4];
  float mrow[8], lrow[8];
#pragma unroll
  for (int nt = 0; nt < 4; ++nt)
#pragma unroll
    for (int e = 0; e < 8; ++e) oacc[nt][e] = 0.0f;
#pragma unroll
  for (int r = 0; r < 8; ++r) { mrow[r] = -1e30f; lrow[r] = 0.0f; }

  const _Float16* kbase = Kb + (size_t)bh * SEQ * DHEAD;   // [j][d]
  const _Float16* vbase = Vb + (size_t)bh * DHEAD * SEQ;   // [d][l]

  for (int j0 = 0; j0 < SEQ; j0 += 32) {
    __syncthreads();
    {  // stage K tile [32j x 64d], 32B vector copies
      int idx = tid * 16, j = idx >> 6, d = idx & 63;
      const uint4* src = (const uint4*)(kbase + ((size_t)(j0 + j)) * DHEAD + d);
      uint4* dst = (uint4*)(sK + j * 64 + d);
      dst[0] = src[0];
      dst[1] = src[1];
    }
    {  // stage V tile [64d x 32j], 32B vector copies (V already transposed)
      int idx = tid * 16, d = idx >> 5, js = idx & 31;
      const uint4* src = (const uint4*)(vbase + (size_t)d * SEQ + j0 + js);
      uint4* dst = (uint4*)(sVT + d * 32 + js);
      dst[0] = src[0];
      dst[1] = src[1];
    }
    if (j0 + 32 < SEQ) {  // prefetch next tiles (global_prefetch_b8)
      __builtin_prefetch(kbase + ((size_t)(j0 + 32)) * DHEAD, 0, 1);
      __builtin_prefetch(vbase + (size_t)(tid >> 1) * SEQ + j0 + 32, 0, 1);
    }
    __syncthreads();

    // S tiles: two 16x16 column blocks, K-dim = 64 (2 WMMAs each)
    v8f s[2];
#pragma unroll
    for (int jt = 0; jt < 2; ++jt) {
      v8f c;
#pragma unroll
      for (int e = 0; e < 8; ++e) c[e] = 0.0f;
#pragma unroll
      for (int kc = 0; kc < 2; ++kc) {
        // B[k=d][n=j] = K[j][d] -> table-order fragment from sK[j][32kc ..]
        v16h bf = load_b_frag(sK + 32 * kc, 64, jt * 16 + ln, g);
        c = wmma_f16(qa[kc], bf, c);
      }
      s[jt] = c;
    }

    // Online softmax: row r (=r+8g) stats reduced across the 16-lane half.
#pragma unroll
    for (int r = 0; r < 8; ++r) {
      float mx = fmaxf(s[0][r], s[1][r]);
#pragma unroll
      for (int msk = 8; msk >= 1; msk >>= 1) mx = fmaxf(mx, __shfl_xor(mx, msk, 32));
      const float mn = fmaxf(mrow[r], mx);
      const float sc = __expf(mrow[r] - mn);
      const float p0 = __expf(s[0][r] - mn);
      const float p1 = __expf(s[1][r] - mn);
      float rs = p0 + p1;
#pragma unroll
      for (int msk = 8; msk >= 1; msk >>= 1) rs += __shfl_xor(rs, msk, 32);
      lrow[r] = lrow[r] * sc + rs;
      mrow[r] = mn;
#pragma unroll
      for (int nt = 0; nt < 4; ++nt) oacc[nt][r] *= sc;
      // pair-packed store: (P[i][j=ln], P[i][j=16+ln]) as one b32
      h2 hv = {(_Float16)p0, (_Float16)p1};
      *(h2*)(&sP[w][(r + 8 * g) * 32 + 2 * ln]) = hv;
    }
    __syncthreads();

    // P A-fragment: row ln, pairs j = 8g..8g+7 -> one contiguous 32B LDS load,
    // then in-lane deinterleave (even halves -> e<8, odd halves -> e>=8).
    v16h t = *(const v16h*)(&sP[w][ln * 32 + 16 * g]);
    v16h pa;
#pragma unroll
    for (int e = 0; e < 8; ++e) { pa[e] = t[2 * e]; pa[8 + e] = t[2 * e + 1]; }

    // O += P(16x32) x V(32x64): 4 WMMAs over d chunks
#pragma unroll
    for (int nt = 0; nt < 4; ++nt) {
      v16h bf = load_b_frag(sVT, 32, nt * 16 + ln, g);
      oacc[nt] = wmma_f16(pa, bf, oacc[nt]);
    }
  }

  // Normalize and store f16 [bh][i][d] (contiguous across lanes in d)
  _Float16* obase = Ob + ((size_t)bh * SEQ + i0) * DHEAD;
#pragma unroll
  for (int nt = 0; nt < 4; ++nt)
#pragma unroll
    for (int r = 0; r < 8; ++r)
      obase[(size_t)(r + 8 * g) * DHEAD + nt * 16 + ln] =
          (_Float16)(oacc[nt][r] / lrow[r]);
}

// ---------------------------------------------------------------------------
// Kernel 3: output projection.  out[b,o,l] = Wout[o,:] . AO[b,:,l] + bias.
// AO lives as f16 Ob[b][h][l][d]; since the contraction index c = h*64+d is
// the fastest axis there, B fragments are two 8-half runs (table k order) straight from
// global memory — no LDS transpose staging at all.
// ---------------------------------------------------------------------------
__global__ __launch_bounds__(128) void out_proj_kernel(
    const float* __restrict__ W, const float* __restrict__ Bias,
    const _Float16* __restrict__ AO, float* __restrict__ Out) {
  const int b  = blockIdx.z;
  const int o0 = blockIdx.y * 64;
  const int l0 = blockIdx.x * 64;
  const int tid  = threadIdx.x;
  const int lane = tid & 31, w = tid >> 5;
  const int g = lane >> 4, ln = lane & 15;

  __shared__ _Float16 sW[64 * 32];   // [o_local][c_local]

  v8f acc[4];
#pragma unroll
  for (int nt = 0; nt < 4; ++nt)
#pragma unroll
    for (int e = 0; e < 8; ++e) acc[nt][e] = 0.0f;

  const _Float16* aob = AO + (size_t)b * HEADS * SEQ * DHEAD;

  for (int c0 = 0; c0 < DIM; c0 += 32) {
    __syncthreads();
    {  // stage W_out tile, packed b32 stores
      int idx = tid * 16, o = idx >> 5, cs = idx & 31;
      const float* src = W + (size_t)(o0 + o) * DIM + c0 + cs;
#pragma unroll
      for (int e = 0; e < 8; ++e) {
        h2 hv = {(_Float16)src[2 * e], (_Float16)src[2 * e + 1]};
        *(h2*)(sW + o * 32 + cs + 2 * e) = hv;
      }
    }
    __syncthreads();
    v16h af = load_a_frag(sW, 32, w * 16 + ln, g, 0);
    // B fragments direct from global (table k order via load_b_frag),
    // h = c0>>6 (uniform within a 32-wide c step).
    const _Float16* bsrc =
        aob + (size_t)(c0 >> 6) * SEQ * DHEAD + (c0 & 63);
#pragma unroll
    for (int nt = 0; nt < 4; ++nt) {
      v16h bf = load_b_frag(bsrc, DHEAD, l0 + nt * 16 + ln, g);
      acc[nt] = wmma_f16(af, bf, acc[nt]);
    }
  }

#pragma unroll
  for (int r = 0; r < 8; ++r) {
    const int o    = o0 + 16 * w + r + 8 * g;
    const float bi = Bias[o];
#pragma unroll
    for (int nt = 0; nt < 4; ++nt) {
      const int l = l0 + nt * 16 + ln;
      Out[((size_t)b * DIM + o) * SEQ + l] = acc[nt][r] + bi;  // coalesced f32
    }
  }
}

// ---------------------------------------------------------------------------
extern "C" void kernel_launch(void* const* d_in, const int* in_sizes, int n_in,
                              void* d_out, int out_size, void* d_ws, size_t ws_size,
                              hipStream_t stream) {
  const float* x     = (const float*)d_in[0];  // (4,512,2048)
  const float* w_qkv = (const float*)d_in[1];  // (1536,512)
  const float* b_qkv = (const float*)d_in[2];  // (1536,)
  const float* w_out = (const float*)d_in[3];  // (512,512)
  const float* b_out = (const float*)d_in[4];  // (512,)
  float* out = (float*)d_out;                  // (4,512,2048)

  const size_t elems = (size_t)BATCH * HEADS * SEQ * DHEAD;  // 4.19M halves each
  _Float16* Q = (_Float16*)d_ws;
  _Float16* K = Q + elems;
  _Float16* V = K + elems;
  _Float16* O = V + elems;  // total 32 MiB of workspace

  qkv_proj_kernel<<<dim3(SEQ / 64, OQKV / 64, BATCH), 128, 0, stream>>>(
      x, w_qkv, b_qkv, Q, K, V);
  flash_attn_kernel<<<dim3(SEQ / 64, HEADS, BATCH), 128, 0, stream>>>(Q, K, V, O);
  out_proj_kernel<<<dim3(SEQ / 64, DIM / 64, BATCH), 128, 0, stream>>>(
      w_out, b_out, O, out);
}
